// PointNetSimpleVn_38783554683623
// MI455X (gfx1250) — matrix operand order repaired and checked by reading, not compiled, not run on hardware
//
#include <hip/hip_runtime.h>

// VN-PointNet (3 layers) for MI455X / gfx1250, wave32 + WMMA f16 (f32 accum)
// + CDNA5 async global->LDS copies for all tile staging.
//
// Pipeline per kernel_launch:
//   1) cvt kernels: all 15 weight tensors f32 -> f16 into d_ws (L2-hot forever).
//   2) knn_kernel: brute-force top-24 per point (async-LDS-tiled candidates,
//      register top-K with tracked-worst replacement).
//   3) vn_layer_kernel<CIN,COUT> x3: fully fused edge-conv. One wave per WG,
//      2 nodes (48 edges = 144 rows) per WG. Entire t1 -> lrelu -> t2 -> dm
//      -> dot -> argmax/gather -> outer lrelu chain lives in dynamic LDS.
//      Dense math via v_wmma_f32_16x16x32_f16 (K=64/128 -> 2/4 k-steps).
//      Each layer additionally emits an f16 copy of its output in
//      [node][dim][C] order so the next layer's A tiles are gathered with
//      global_load_async_to_lds_b128 (ASYNCcnt) directly into WMMA row order.
//
// d_ws layout: [0, ~222KB) f16 weights; [256KB, +960KB) knn indices;
//              [2MB, +3.84MB) aux f16 of h1; [6MB, +3.84MB) aux f16 of h2.

#define NP   10000
#define KNN  24
#define VN_EPS 1e-7f

#define NODES_PER_WG 2
#define E_WG  (NODES_PER_WG * KNN)   // 48 edges
#define ROWS  (E_WG * 3)             // 144 GEMM rows

typedef __attribute__((ext_vector_type(16))) _Float16 v16h;
typedef __attribute__((ext_vector_type(8)))  float    v8f;

// Low 32 bits of a shared-aperture flat address == LDS byte offset (ISA 10.2).
static __device__ __forceinline__ unsigned lds_off_of(const void* p) {
    return (unsigned)(unsigned long long)p;
}
static __device__ __forceinline__ void async_b128(unsigned lds_off, const void* g) {
    unsigned long long ga = (unsigned long long)g;
    asm volatile("global_load_async_to_lds_b128 %0, %1, off"
                 :: "v"(lds_off), "v"(ga) : "memory");
}
static __device__ __forceinline__ void async_b32(unsigned lds_off, const void* g, int /*unused*/) {
    unsigned long long ga = (unsigned long long)g;
    asm volatile("global_load_async_to_lds_b32 %0, %1, off"
                 :: "v"(lds_off), "v"(ga) : "memory");
}
static __device__ __forceinline__ void wait_async0() {
    asm volatile("s_wait_asynccnt 0x0" ::: "memory");
}

// ---------------------------------------------------------------------------
// f32 -> f16 weight conversion (one-time prep into d_ws)
// ---------------------------------------------------------------------------
__global__ void __launch_bounds__(256)
cvt_f32_f16_kernel(const float* __restrict__ s, _Float16* __restrict__ d, int n) {
    int i = blockIdx.x * 256 + threadIdx.x;
    if (i < n) d[i] = (_Float16)s[i];
}

// ---------------------------------------------------------------------------
// Brute-force KNN: one query per thread; candidate tiles staged into LDS with
// async global->LDS b32 copies; top-K in registers, tracked worst slot.
// ---------------------------------------------------------------------------
__global__ void __launch_bounds__(256)
knn_kernel(const float* __restrict__ pos, int* __restrict__ knn_out) {
    __shared__ float cp[256 * 3];
    const int gid = blockIdx.x * 256 + threadIdx.x;
    const bool valid = gid < NP;
    float qx = 0.f, qy = 0.f, qz = 0.f;
    if (valid) { qx = pos[gid*3+0]; qy = pos[gid*3+1]; qz = pos[gid*3+2]; }

    float bd[KNN]; int bi[KNN];
#pragma unroll
    for (int k = 0; k < KNN; ++k) { bd[k] = 3.4e38f; bi[k] = 0; }
    float worst = 3.4e38f; int wslot = 0;

    for (int ts = 0; ts < NP; ts += 256) {
        const int cnt = (NP - ts < 256) ? (NP - ts) : 256;
        if ((int)threadIdx.x < cnt) {
            const float* g = pos + (size_t)(ts + threadIdx.x) * 3;
            const unsigned l = lds_off_of(&cp[threadIdx.x * 3]);
            async_b32(l,     g,     0);
            async_b32(l + 4, g + 1, 0);
            async_b32(l + 8, g + 2, 0);
        }
        wait_async0();
        __syncthreads();
        if (valid) {
            for (int j = 0; j < cnt; ++j) {
                const float dx = qx - cp[j*3+0];
                const float dy = qy - cp[j*3+1];
                const float dz = qz - cp[j*3+2];
                const float dist = dx*dx + dy*dy + dz*dz;
                if (dist < worst) {
#pragma unroll
                    for (int k = 0; k < KNN; ++k)
                        if (k == wslot) { bd[k] = dist; bi[k] = ts + j; }
                    worst = -1.f;
#pragma unroll
                    for (int k = 0; k < KNN; ++k)
                        if (bd[k] > worst) { worst = bd[k]; wslot = k; }
                }
            }
        }
        __syncthreads();
    }
    if (valid) {
#pragma unroll
        for (int k = 0; k < KNN; ++k) knn_out[gid*KNN + k] = bi[k];
    }
}

// ---------------------------------------------------------------------------
// WMMA GEMM over the workgroup's 144 x KDIM tile (A in LDS, f16) times a
// KDIM x COUT weight (B straight from global f16: lane n's column == W[n,:]).
// Epilogue optionally adds the relative-position rank-1 term (W1 GEMM) and
// stores f16 into O (LDS).
// ---------------------------------------------------------------------------
template <int COUT, int KDIM, bool ADD_REL>
__device__ inline void gemm_all(const _Float16* __restrict__ A, int a_stride,
                                const _Float16* __restrict__ B, int b_stride,
                                _Float16* __restrict__ O,
                                const float* __restrict__ w1_f32, int w1_stride,
                                const float* __restrict__ rel_s, int lane) {
    constexpr int MT = ROWS / 16;      // 9 row tiles
    constexpr int NT = COUT / 16;      // 4 or 8 col tiles
    constexpr int KS = KDIM / 32;      // 2 or 4 k-steps
    const int m15 = lane & 15;
    const int khalf = lane >> 4;

    for (int nt = 0; nt < NT; ++nt) {
        const int n = nt * 16 + m15;
        // Hide L2 latency of the next B panel.
        if (nt + 1 < NT) __builtin_prefetch(B + (n + 16) * b_stride, 0, 1);

        v8f acc[MT];
#pragma unroll
        for (int mt = 0; mt < MT; ++mt) { v8f z = {}; acc[mt] = z; }

#pragma unroll
        for (int kt = 0; kt < KS; ++kt) {
            const int kb = kt * 32;
            // B fragment: column n, K = kb + khalf*16 .. +15 (contiguous in W row)
            v16h b;
            const _Float16* bp = B + n * b_stride + kb + khalf * 16;
#pragma unroll
            for (int i = 0; i < 16; ++i) b[i] = bp[i];

#pragma unroll
            for (int mt = 0; mt < MT; ++mt) {
                // A fragment: row m, K = kb + khalf*16 .. +15 (contiguous)
                v16h a;
                const _Float16* ap = A + (mt*16 + m15) * a_stride + kb + khalf * 16;
#pragma unroll
                for (int i = 0; i < 16; ++i) a[i] = ap[i];
                acc[mt] = __builtin_amdgcn_wmma_f32_16x16x32_f16(
                    false, a, false, b, (short)0, acc[mt], false, false);
            }
        }
        // Epilogue: D layout m = r + 8*khalf, n = lane&15
        float w1r = 0.f;
        if constexpr (ADD_REL) w1r = w1_f32[n * w1_stride + (w1_stride - 1)];
#pragma unroll
        for (int mt = 0; mt < MT; ++mt) {
#pragma unroll
            for (int r = 0; r < 8; ++r) {
                const int row = mt*16 + r + 8*khalf;
                float v = acc[mt][r];
                if constexpr (ADD_REL) v += w1r * rel_s[row];
                O[row * COUT + n] = (_Float16)v;
            }
        }
    }
}

// ---------------------------------------------------------------------------
// Fused per-layer edge-conv. blockDim = 32 (one wave). 2 nodes per WG.
// aux_in:  f16 [N][3][CIN_X] copy of previous layer output (layers 2/3).
// aux_out: f16 [N][3][COUT]  copy of this layer's output (null for layer 3).
// ---------------------------------------------------------------------------
template <int CIN_X, int COUT>
__global__ void __launch_bounds__(32)
vn_layer_kernel(const _Float16* __restrict__ aux_in,
                const float* __restrict__ pos,      // [N, 3]
                const int*   __restrict__ knn,      // [N, KNN]
                const float* __restrict__ w1_f32,   // [COUT, CIN_X+1]
                const _Float16* __restrict__ w1_h,  // f16 copy of w1
                const _Float16* __restrict__ wd_h,  // [COUT, COUT]
                const _Float16* __restrict__ w2_h,
                const _Float16* __restrict__ wm_h,
                const _Float16* __restrict__ rw_h,
                float* __restrict__ out,            // [N, COUT, 3]
                _Float16* __restrict__ aux_out) {   // [N, 3, COUT] or null
    constexpr int XSE = (CIN_X > 1) ? ROWS * CIN_X : 2;
    extern __shared__ char smem[];
    _Float16* bufA  = (_Float16*)smem;                 // [ROWS][COUT]
    _Float16* bufB  = bufA + ROWS * COUT;              // [ROWS][COUT]
    _Float16* xs    = bufB + ROWS * COUT;              // [ROWS][CIN_X] (layers 2/3)
    float* rel_s    = (float*)(xs + XSE);              // [ROWS]
    float* psrc_s   = rel_s + ROWS;                    // [ROWS] (layer 1)
    float* dot2     = psrc_s + ROWS;                   // [E_WG][COUT]
    float* ys       = dot2 + E_WG * COUT;              // [2][COUT][3]
    int*   src_s    = (int*)(ys + 2 * COUT * 3);       // [E_WG]

    const int lane  = threadIdx.x;
    const int node0 = blockIdx.x * NODES_PER_WG;

    // P0: neighbor indices
    for (int i = lane; i < E_WG; i += 32) src_s[i] = knn[node0 * KNN + i];
    __syncthreads();

    // P0b: relative positions (and source pos for layer 1)
    for (int r = lane; r < ROWS; r += 32) {
        const int e = r / 3, d = r % 3;
        const int s = src_s[e];
        const int dn = node0 + e / KNN;
        const float ps = pos[s*3 + d];
        rel_s[r] = ps - pos[dn*3 + d];
        if constexpr (CIN_X == 1) psrc_s[r] = ps;
    }
    // P0c: async-gather aux_in rows straight into WMMA A-row order.
    // Row (e,d) of A == aux_in[src*3 + d][0..CIN_X) : contiguous f16 run.
    if constexpr (CIN_X > 1) {
        constexpr int CHUNKS = (CIN_X * 2) / 16;         // 16B chunks per row (8)
        constexpr int RPI = 32 / CHUNKS;                 // rows per wave-iter (4)
        const int rsub = lane / CHUNKS;
        const int c    = lane % CHUNKS;
        for (int rb = 0; rb < ROWS; rb += RPI) {
            const int r = rb + rsub;
            const int e = r / 3, d = r % 3;
            const int s = src_s[e];
            const void* g = (const void*)((const char*)aux_in +
                            ((size_t)(s*3 + d) * CIN_X) * 2 + (size_t)c * 16);
            async_b128(lds_off_of(xs) + (unsigned)(r * CIN_X * 2 + c * 16), g);
        }
        wait_async0();
    }
    __syncthreads();

    // P1: t1 = VNLinear(h_in, W1)  (rel channel folded as rank-1 epilogue)
    if constexpr (CIN_X == 1) {
        for (int t = lane; t < ROWS * COUT; t += 32) {
            const int row = t / COUT, o = t % COUT;
            const float v = w1_f32[o*2+0] * psrc_s[row] + w1_f32[o*2+1] * rel_s[row];
            bufA[row * COUT + o] = (_Float16)v;
        }
    } else {
        gemm_all<COUT, CIN_X, true>(xs, CIN_X, w1_h, CIN_X + 1, bufA,
                                    w1_f32, CIN_X + 1, rel_s, lane);
    }
    __syncthreads();

    // P2: d = VNLinear(t1, Wd)
    gemm_all<COUT, COUT, false>(bufA, COUT, wd_h, COUT, bufB,
                                nullptr, 0, nullptr, lane);
    __syncthreads();

    // P3: VN-LeakyReLU(slope 0): h = mask*x + (1-mask)*(x - dot/(|d|^2+eps)*d)
    for (int t = lane; t < E_WG * COUT; t += 32) {
        const int e = t / COUT, o = t % COUT;
        float x0 = (float)bufA[(e*3+0)*COUT + o];
        float x1 = (float)bufA[(e*3+1)*COUT + o];
        float x2 = (float)bufA[(e*3+2)*COUT + o];
        const float d0 = (float)bufB[(e*3+0)*COUT + o];
        const float d1 = (float)bufB[(e*3+1)*COUT + o];
        const float d2 = (float)bufB[(e*3+2)*COUT + o];
        const float dot = x0*d0 + x1*d1 + x2*d2;
        if (dot < 0.f) {
            const float s = dot / (d0*d0 + d1*d1 + d2*d2 + VN_EPS);
            x0 -= s*d0; x1 -= s*d1; x2 -= s*d2;
        }
        bufA[(e*3+0)*COUT + o] = (_Float16)x0;
        bufA[(e*3+1)*COUT + o] = (_Float16)x1;
        bufA[(e*3+2)*COUT + o] = (_Float16)x2;
    }
    __syncthreads();

    // P4: t2 = VNLinear(h, W2)
    gemm_all<COUT, COUT, false>(bufA, COUT, w2_h, COUT, bufB,
                                nullptr, 0, nullptr, lane);
    __syncthreads();

    // P5: dm = VNLinear(t2, Wm)
    gemm_all<COUT, COUT, false>(bufB, COUT, wm_h, COUT, bufA,
                                nullptr, 0, nullptr, lane);
    __syncthreads();

    // P6: dot2[e][o] = sum_d t2*dm
    for (int t = lane; t < E_WG * COUT; t += 32) {
        const int e = t / COUT, o = t % COUT;
        float s = 0.f;
#pragma unroll
        for (int d = 0; d < 3; ++d)
            s += (float)bufB[(e*3+d)*COUT + o] * (float)bufA[(e*3+d)*COUT + o];
        dot2[e * COUT + o] = s;
    }
    __syncthreads();

    // P7: per-node argmax over K, gather t2 -> ys
    for (int t = lane; t < NODES_PER_WG * COUT; t += 32) {
        const int nd = t / COUT, o = t % COUT;
        float best = -3.4e38f; int bk = 0;
        for (int k = 0; k < KNN; ++k) {
            const float v = dot2[(nd*KNN + k) * COUT + o];
            if (v > best) { best = v; bk = k; }
        }
#pragma unroll
        for (int d = 0; d < 3; ++d)
            ys[(nd*COUT + o)*3 + d] = (float)bufB[((nd*KNN + bk)*3 + d)*COUT + o];
    }
    __syncthreads();

    // P8: outer VN-LeakyReLU with Rw, write f32 output (+ f16 aux copy)
    for (int t = lane; t < NODES_PER_WG * COUT; t += 32) {
        const int nd = t / COUT, o = t % COUT;
        float dv0 = 0.f, dv1 = 0.f, dv2 = 0.f;
        for (int i = 0; i < COUT; ++i) {
            const float w = (float)rw_h[o*COUT + i];
            dv0 += ys[(nd*COUT + i)*3 + 0] * w;
            dv1 += ys[(nd*COUT + i)*3 + 1] * w;
            dv2 += ys[(nd*COUT + i)*3 + 2] * w;
        }
        float y0 = ys[(nd*COUT + o)*3 + 0];
        float y1 = ys[(nd*COUT + o)*3 + 1];
        float y2 = ys[(nd*COUT + o)*3 + 2];
        const float dot = y0*dv0 + y1*dv1 + y2*dv2;
        if (dot < 0.f) {
            const float s = dot / (dv0*dv0 + dv1*dv1 + dv2*dv2 + VN_EPS);
            y0 -= s*dv0; y1 -= s*dv1; y2 -= s*dv2;
        }
        const int node = node0 + nd;
        const int nb = node * COUT * 3 + o * 3;
        out[nb+0] = y0; out[nb+1] = y1; out[nb+2] = y2;
        if (aux_out) {
            aux_out[((size_t)node*3 + 0) * COUT + o] = (_Float16)y0;
            aux_out[((size_t)node*3 + 1) * COUT + o] = (_Float16)y1;
            aux_out[((size_t)node*3 + 2) * COUT + o] = (_Float16)y2;
        }
    }
}

// ---------------------------------------------------------------------------
static constexpr int WSZ[15] = {
    128, 4096, 4096, 4096, 4096,        // c1_w1, c1_wd, c1_w2, c1_wm, r1_w
    4160, 4096, 4096, 4096, 4096,       // c2_*
    8320, 16384, 16384, 16384, 16384    // c3_*
};

static constexpr size_t layer_smem(int cin_x, int cout) {
    const int xse = (cin_x > 1) ? ROWS * cin_x : 2;
    return (size_t)(2 * ROWS * cout) * 2       // bufA + bufB (f16)
         + (size_t)xse * 2                     // xs (f16)
         + (size_t)ROWS * 4 * 2                // rel_s + psrc_s
         + (size_t)E_WG * cout * 4             // dot2
         + (size_t)2 * cout * 3 * 4            // ys
         + (size_t)E_WG * 4;                   // src_s
}

extern "C" void kernel_launch(void* const* d_in, const int* in_sizes, int n_in,
                              void* d_out, int out_size, void* d_ws, size_t ws_size,
                              hipStream_t stream) {
    (void)in_sizes; (void)n_in; (void)out_size; (void)ws_size;

    const float* pos = (const float*)d_in[0];
    _Float16* wh = (_Float16*)d_ws;

    int off[15]; { int a = 0; for (int i = 0; i < 15; ++i) { off[i] = a; a += WSZ[i]; } }

    // 1) weights f32 -> f16 into ws
    for (int i = 0; i < 15; ++i) {
        const int n = WSZ[i];
        cvt_f32_f16_kernel<<<(n + 255) / 256, 256, 0, stream>>>(
            (const float*)d_in[1 + i], wh + off[i], n);
    }

    // 2) knn indices at 256KB into ws
    int* knn = (int*)((char*)d_ws + (256u * 1024u));
    knn_kernel<<<(NP + 255) / 256, 256, 0, stream>>>(pos, knn);

    // aux f16 copies of layer outputs (A-gather sources for the next layer)
    _Float16* aux1 = (_Float16*)((char*)d_ws + (2u  * 1024u * 1024u)); // [N][3][64]
    _Float16* aux2 = (_Float16*)((char*)d_ws + (6u  * 1024u * 1024u)); // [N][3][64]

    // 3) fused layers
    float* h1 = (float*)d_out;
    float* h2 = h1 + (size_t)NP * 64 * 3;
    float* h3 = h2 + (size_t)NP * 64 * 3;

    vn_layer_kernel<1, 64><<<NP / NODES_PER_WG, 32, layer_smem(1, 64), stream>>>(
        nullptr, pos, knn, (const float*)d_in[1],
        wh + off[0], wh + off[1], wh + off[2], wh + off[3], wh + off[4], h1, aux1);

    vn_layer_kernel<64, 64><<<NP / NODES_PER_WG, 32, layer_smem(64, 64), stream>>>(
        aux1, pos, knn, (const float*)d_in[6],
        wh + off[5], wh + off[6], wh + off[7], wh + off[8], wh + off[9], h2, aux2);

    vn_layer_kernel<64, 128><<<NP / NODES_PER_WG, 32, layer_smem(64, 128), stream>>>(
        aux2, pos, knn, (const float*)d_in[11],
        wh + off[10], wh + off[11], wh + off[12], wh + off[13], wh + off[14], h3, nullptr);
}
